// DiscoNetFusion_83597243450081
// MI455X (gfx1250) — matrix operand order repaired and checked by reading, not compiled, not run on hardware
//
#include <hip/hip_runtime.h>

// DiscoNet fusion, fully fused single kernel for MI455X (gfx1250, wave32).
// Per wave: 16 pixels. Lanes 0-15 = warped-neighbor half of the WMMA B operand
// (K=0..15), lanes 16-31 = ego half (K=16..31). Conv tower 32->16->8->4 runs as
// three v_wmma_f32_16x16x32_f16 with BN folded into A; conv4/softmax/fusion/
// final 16x16 conv are VALU on lanes 0-15. No LDS, no workspace, no barriers.

#define LN 5
#define CN 16
#define HI_ 256
#define WI_ 256
#define HW (HI_ * WI_)
#define BN_EPS 1e-5f
#define INV_DS 1.25f // 1/(0.4*2)

typedef _Float16 v16h __attribute__((ext_vector_type(16)));
typedef float    v8f  __attribute__((ext_vector_type(8)));

__device__ __forceinline__ float swap16(float v) {
    // ds_swizzle SWAPX16: and=0x1f, or=0, xor=0x10 -> 0x401F
    int iv = __builtin_bit_cast(int, v);
    iv = __builtin_amdgcn_ds_swizzle(iv, 0x401F);
    return __builtin_bit_cast(float, iv);
}

__global__ __launch_bounds__(256, 1) void disconet_fused(
    const float* __restrict__ x,   const float* __restrict__ ptm,
    const float* __restrict__ w1,  const float* __restrict__ b1,
    const float* __restrict__ g1,  const float* __restrict__ be1,
    const float* __restrict__ rm1, const float* __restrict__ rv1,
    const float* __restrict__ w2,  const float* __restrict__ b2,
    const float* __restrict__ g2,  const float* __restrict__ be2,
    const float* __restrict__ rm2, const float* __restrict__ rv2,
    const float* __restrict__ w3,  const float* __restrict__ b3,
    const float* __restrict__ g3,  const float* __restrict__ be3,
    const float* __restrict__ rm3, const float* __restrict__ rv3,
    const float* __restrict__ w4,  const float* __restrict__ b4,
    const float* __restrict__ wm,  const float* __restrict__ bm,
    float* __restrict__ out)
{
    const int i    = blockIdx.y;          // ego frame
    const int lane = threadIdx.x & 31;
    const int wv   = threadIdx.x >> 5;
    const int n    = lane & 15;           // WMMA N (pixel within tile)
    const int hi   = lane >> 4;           // 0: nb half (K0..15), 1: ego half (K16..31)
    const int m    = n;                   // WMMA A row for this lane

    const int pix = ((blockIdx.x * 8 + wv) << 4) + n;   // W=256 multiple of 16
    const int py  = pix >> 8;
    const int px  = pix & 255;

    // ---------------- fold BN scale into WMMA A operands (f16) ----------------
    // 16-bit A layout (contiguous K): lane m holds K 0..15; lane m+16 holds K 16..31.
    v16h A1, A2, A3;
    {
        const float s1 = g1[m] * rsqrtf(rv1[m] + BN_EPS);
        const float s2 = (m < 8) ? g2[m] * rsqrtf(rv2[m] + BN_EPS) : 0.f;
        const float s3 = (m < 4) ? g3[m] * rsqrtf(rv3[m] + BN_EPS) : 0.f;
#pragma unroll
        for (int e = 0; e < 16; ++e) {
            const int k = e + hi * 16;      // K index for element e
            A1[e] = (_Float16)(w1[m * 32 + k] * s1);
            A2[e] = (_Float16)((m < 8 && k < 16) ? w2[m * 16 + k] * s2 : 0.f);
            A3[e] = (_Float16)((m < 4 && k < 8 ) ? w3[m * 8  + k] * s3 : 0.f);
        }
    }

    // Folded BN biases per accumulator row (D row r on this lane = oc r+8*hi)
    float bias1[8], bias2[8], bias3[8];
#pragma unroll
    for (int r = 0; r < 8; ++r) {
        const int oc = r + hi * 8;
        bias1[r] = (b1[oc] - rm1[oc]) * (g1[oc] * rsqrtf(rv1[oc] + BN_EPS)) + be1[oc];
        bias2[r] = (!hi)           ? (b2[r] - rm2[r]) * (g2[r] * rsqrtf(rv2[r] + BN_EPS)) + be2[r] : 0.f;
        bias3[r] = (!hi && r < 4)  ? (b3[r] - rm3[r]) * (g3[r] * rsqrtf(rv3[r] + BN_EPS)) + be3[r] : 0.f;
    }
    const float w4r0 = w4[0], w4r1 = w4[1], w4r2 = w4[2], w4r3 = w4[3];
    const float b4r  = b4[0];

    // ---------------- gather: warped neighbors (lanes 0-15) / ego (lanes 16-31) ----
    v16h nb16[LN];   // doubles as the WMMA B operand for conv1 (K0..15 nb | K16..31 ego)
    if (hi) {
        const float* e0 = x + (size_t)i * CN * HW + py * WI_ + px;
#pragma unroll
        for (int c = 0; c < 16; ++c) {
            const _Float16 h = (_Float16)e0[c * HW];
#pragma unroll
            for (int j = 0; j < LN; ++j) nb16[j][c] = h;
        }
    } else {
#pragma unroll
        for (int j = 0; j < LN; ++j) {
            // affine T[j][i]; inverse of [[a,b,tx],[c,d,ty],[0,0,1]]
            const float* mp = ptm + ((j * LN) + i) * 16;
            const float a  = mp[0], bb = mp[1], tx = mp[3] * INV_DS;
            const float cc = mp[4], dd = mp[5], ty = mp[7] * INV_DS;
            const float rdet = 1.f / (a * dd - bb * cc);
            const float ia =  dd * rdet, ib = -bb * rdet;
            const float ic = -cc * rdet, id =  a  * rdet;
            const float sx = ia * (float)px + ib * (float)py - (ia * tx + ib * ty);
            const float sy = ic * (float)px + id * (float)py - (ic * tx + id * ty);

            const float x0 = floorf(sx), y0 = floorf(sy);
            const float fx = sx - x0,   fy = sy - y0;
            const float x1 = x0 + 1.f,  y1 = y0 + 1.f;
            const bool vx0 = (x0 >= 0.f) && (x0 <= (float)(WI_ - 1));
            const bool vx1 = (x1 >= 0.f) && (x1 <= (float)(WI_ - 1));
            const bool vy0 = (y0 >= 0.f) && (y0 <= (float)(HI_ - 1));
            const bool vy1 = (y1 >= 0.f) && (y1 <= (float)(HI_ - 1));
            const float w00 = (vx0 && vy0) ? (1.f - fx) * (1.f - fy) : 0.f;
            const float w10 = (vx1 && vy0) ? fx * (1.f - fy)         : 0.f;
            const float w01 = (vx0 && vy1) ? (1.f - fx) * fy         : 0.f;
            const float w11 = (vx1 && vy1) ? fx * fy                 : 0.f;
            const int ix0 = (int)fminf(fmaxf(x0, 0.f), (float)(WI_ - 1));
            const int ix1 = (int)fminf(fmaxf(x1, 0.f), (float)(WI_ - 1));
            const int iy0 = (int)fminf(fmaxf(y0, 0.f), (float)(HI_ - 1));
            const int iy1 = (int)fminf(fmaxf(y1, 0.f), (float)(HI_ - 1));
            const int o00 = iy0 * WI_ + ix0, o10 = iy0 * WI_ + ix1;
            const int o01 = iy1 * WI_ + ix0, o11 = iy1 * WI_ + ix1;
            const float* bj = x + (size_t)j * CN * HW;
#pragma unroll
            for (int c = 0; c < 16; ++c) {
                const float* bc = bj + c * HW;
                const float v = bc[o00] * w00 + bc[o10] * w10 + bc[o01] * w01 + bc[o11] * w11;
                nb16[j][c] = (_Float16)v;
            }
        }
    }

    // ---------------- conv tower (WMMA) + logits, per neighbor j ----------------
    float logits[LN];
    const v8f zero = {0.f, 0.f, 0.f, 0.f, 0.f, 0.f, 0.f, 0.f};
#pragma unroll
    for (int j = 0; j < LN; ++j) {
        // conv1: 32 -> 16
        v8f d1 = __builtin_amdgcn_wmma_f32_16x16x32_f16(
            false, A1, false, nb16[j], (short)0, zero, false, false);
        float y1[8], pr[8];
#pragma unroll
        for (int r = 0; r < 8; ++r) y1[r] = fmaxf(d1[r] + bias1[r], 0.f);
#pragma unroll
        for (int r = 0; r < 8; ++r) pr[r] = swap16(y1[r]);   // partner rows 8..15
        v16h B2;
#pragma unroll
        for (int e = 0; e < 16; ++e) {
            const float v = hi ? 0.f : (e < 8 ? y1[e] : pr[e - 8]);
            B2[e] = (_Float16)v;
        }
        // conv2: 16 -> 8
        v8f d2 = __builtin_amdgcn_wmma_f32_16x16x32_f16(
            false, A2, false, B2, (short)0, zero, false, false);
        float y2[8];
#pragma unroll
        for (int r = 0; r < 8; ++r) y2[r] = fmaxf(d2[r] + bias2[r], 0.f);
        v16h B3;
#pragma unroll
        for (int e = 0; e < 16; ++e)
            B3[e] = (_Float16)((!hi && e < 8) ? y2[e] : 0.f);
        // conv3: 8 -> 4
        v8f d3 = __builtin_amdgcn_wmma_f32_16x16x32_f16(
            false, A3, false, B3, (short)0, zero, false, false);
        // conv4: 4 -> 1 (+ReLU)
        float lg = b4r;
        lg += w4r0 * fmaxf(d3[0] + bias3[0], 0.f);
        lg += w4r1 * fmaxf(d3[1] + bias3[1], 0.f);
        lg += w4r2 * fmaxf(d3[2] + bias3[2], 0.f);
        lg += w4r3 * fmaxf(d3[3] + bias3[3], 0.f);
        logits[j] = fmaxf(lg, 0.f);
    }

    // ---------------- softmax over j, weighted fusion, final 16x16 conv ----------
    if (!hi) {
        float mx = logits[0];
#pragma unroll
        for (int j = 1; j < LN; ++j) mx = fmaxf(mx, logits[j]);
        float wj[LN], s = 0.f;
#pragma unroll
        for (int j = 0; j < LN; ++j) { wj[j] = __expf(logits[j] - mx); s += wj[j]; }
        const float inv = 1.f / s;

        float fused[16];
#pragma unroll
        for (int c = 0; c < 16; ++c) {
            float acc = 0.f;
#pragma unroll
            for (int j = 0; j < LN; ++j) acc += wj[j] * (float)nb16[j][c];
            fused[c] = acc * inv;
        }

        float* op = out + (size_t)i * CN * HW + py * WI_ + px;
#pragma unroll
        for (int o = 0; o < 16; ++o) {
            float acc = bm[o];
#pragma unroll
            for (int c = 0; c < 16; ++c) acc += wm[o * 16 + c] * fused[c];
            op[o * HW] = acc;
        }
    }
}

extern "C" void kernel_launch(void* const* d_in, const int* in_sizes, int n_in,
                              void* d_out, int out_size, void* d_ws, size_t ws_size,
                              hipStream_t stream) {
    (void)in_sizes; (void)n_in; (void)d_ws; (void)ws_size; (void)out_size;
    const float* x   = (const float*)d_in[0];
    // d_in[1] = record_len (int, unused)
    const float* ptm = (const float*)d_in[2];
    const float* w1  = (const float*)d_in[3];
    const float* b1  = (const float*)d_in[4];
    const float* g1  = (const float*)d_in[5];
    const float* be1 = (const float*)d_in[6];
    const float* rm1 = (const float*)d_in[7];
    const float* rv1 = (const float*)d_in[8];
    const float* w2  = (const float*)d_in[9];
    const float* b2  = (const float*)d_in[10];
    const float* g2  = (const float*)d_in[11];
    const float* be2 = (const float*)d_in[12];
    const float* rm2 = (const float*)d_in[13];
    const float* rv2 = (const float*)d_in[14];
    const float* w3  = (const float*)d_in[15];
    const float* b3  = (const float*)d_in[16];
    const float* g3  = (const float*)d_in[17];
    const float* be3 = (const float*)d_in[18];
    const float* rm3 = (const float*)d_in[19];
    const float* rv3 = (const float*)d_in[20];
    const float* w4  = (const float*)d_in[21];
    const float* b4  = (const float*)d_in[22];
    const float* wm  = (const float*)d_in[23];
    const float* bm  = (const float*)d_in[24];
    float* out = (float*)d_out;

    // 5 ego frames x (65536 px / 128 px-per-block) = (512, 5) blocks, 256 threads.
    dim3 grid(512, 5);
    disconet_fused<<<grid, 256, 0, stream>>>(
        x, ptm, w1, b1, g1, be1, rm1, rv1, w2, b2, g2, be2, rm2, rv2,
        w3, b3, g3, be3, rm3, rv3, w4, b4, wm, bm, out);
}
